// GeoS4FormerNetV2_34402688041059
// MI455X (gfx1250) — matrix operand order repaired and checked by reading, not compiled, not run on hardware
//
#include <hip/hip_runtime.h>
#include <math.h>

// ---------------- constants from the reference ----------------
#define B_     512
#define L_     512
#define HID_   128
#define NST_   64
#define NNODE_ 50000
#define NEDGE_ 500000
#define GL_    3

typedef __bf16 bf16_t;
typedef __bf16  v16bf __attribute__((ext_vector_type(16)));
typedef float   v8f   __attribute__((ext_vector_type(8)));

// ---------------- device helpers ----------------
__device__ __forceinline__ float gelu_f(float x) {
    const float c = 0.7978845608028654f;
    return 0.5f * x * (1.0f + tanhf(c * (x + 0.044715f * x * x * x)));
}
__device__ __forceinline__ float sigmoid_f(float x) { return 1.0f / (1.0f + __expf(-x)); }

// monotone float<->uint key for atomicMax on signed floats
__device__ __forceinline__ unsigned fkey(float x) {
    unsigned b = __float_as_uint(x);
    return (b & 0x80000000u) ? ~b : (b | 0x80000000u);
}
__device__ __forceinline__ float funkey(unsigned k) {
    unsigned b = (k & 0x80000000u) ? (k ^ 0x80000000u) : ~k;
    return __uint_as_float(b);
}

// blockDim.x == 128 row LayerNorm; red is 128-float LDS scratch
__device__ __forceinline__ float block_ln_128(float t, float g, float bb, float* red, int d) {
    red[d] = t; __syncthreads();
    for (int s = 64; s > 0; s >>= 1) { if (d < s) red[d] += red[d + s]; __syncthreads(); }
    float mu = red[0] * (1.0f / 128.0f); __syncthreads();
    float c = t - mu;
    red[d] = c * c; __syncthreads();
    for (int s = 64; s > 0; s >>= 1) { if (d < s) red[d] += red[d + s]; __syncthreads(); }
    float var = red[0] * (1.0f / 128.0f); __syncthreads();
    return c * rsqrtf(var + 1e-5f) * g + bb;
}

// ---------------- generic bf16 WMMA GEMM: C(MxN) = A(MxK) @ B(KxN) ----------------
// one wave (32 lanes) computes one 16x16 f32 tile; K multiple of 32, M,N multiples of 16.
__global__ void k_gemm_bf16(const bf16_t* __restrict__ A, const bf16_t* __restrict__ Bm,
                            float* __restrict__ C, int M, int N, int K) {
    __shared__ bf16_t As[16 * 32];
    __shared__ bf16_t Bs[32 * 16];
    const int tm = blockIdx.x, tn = blockIdx.y;
    const int lane = threadIdx.x;           // 0..31, wave32
    v8f acc = {};
    for (int k0 = 0; k0 < K; k0 += 32) {
        // cooperative tile load: 512 halfs each, 16 per lane
        #pragma unroll
        for (int i = 0; i < 16; ++i) {
            int idx = lane * 16 + i;
            int ra = idx >> 5, ca = idx & 31;                 // A tile 16x32
            As[idx] = A[(size_t)(tm * 16 + ra) * K + k0 + ca];
            int rb = idx >> 4, cb = idx & 15;                 // B tile 32x16
            Bs[idx] = Bm[(size_t)(k0 + rb) * N + tn * 16 + cb];
        }
        if (k0 + 32 < K)  // hint next A tile into cache (global_prefetch_b8)
            __builtin_prefetch(&A[(size_t)(tm * 16 + (lane & 15)) * K + k0 + 32], 0, 1);
        __syncthreads();
        // assemble fragments per CDNA5 16-bit WMMA VGPR layouts
        v16bf afrag, bfrag;
        #pragma unroll
        for (int i = 0; i < 16; ++i) {
            int ka = i + ((i >= 8) ? 8 : 0) + ((lane >= 16) ? 8 : 0); // A: lanes 0-15 K{0-7,16-23}, lanes 16-31 K{8-15,24-31}
            afrag[i] = As[(lane & 15) * 32 + ka];
            int kb = i + ((i >= 8) ? 8 : 0) + ((lane >= 16) ? 8 : 0); // B: lanes 0-15 K{0-7,16-23}, lanes 16-31 K{8-15,24-31}
            bfrag[i] = Bs[kb * 16 + (lane & 15)];
        }
        acc = __builtin_amdgcn_wmma_f32_16x16x32_bf16(false, afrag, false, bfrag,
                                                      (short)0, acc, false, false);
        __syncthreads();
    }
    // C layout: VGPR r -> M = r + 8*(lane>=16), N = lane&15
    #pragma unroll
    for (int r = 0; r < 8; ++r) {
        int m = tm * 16 + r + ((lane >= 16) ? 8 : 0);
        int n = tn * 16 + (lane & 15);
        C[(size_t)m * N + n] = acc[r];
    }
}

// ---------------- small utility kernels ----------------
__global__ void k_cast_bf16(const float* __restrict__ s, bf16_t* __restrict__ d, int n) {
    int i = blockIdx.x * blockDim.x + threadIdx.x;
    if (i < n) d[i] = (bf16_t)s[i];
}
__global__ void k_fill_u32(unsigned* __restrict__ d, unsigned v, int n) {
    int i = blockIdx.x * blockDim.x + threadIdx.x;
    if (i < n) d[i] = v;
}

// ---------------- S4D kernel computation ----------------
__global__ void k_s4_prep(const float* __restrict__ log_dt, const float* __restrict__ A_re,
                          const float* __restrict__ A_im, const float* __restrict__ C_re,
                          const float* __restrict__ C_im, float* __restrict__ dtA_re,
                          float* __restrict__ dtA_im, float* __restrict__ Ct_re,
                          float* __restrict__ Ct_im) {
    int i = blockIdx.x * blockDim.x + threadIdx.x;   // d*NST + n
    if (i >= HID_ * NST_) return;
    int d = i / NST_;
    float dt = __expf(log_dt[d]);
    float ar = -__expf(A_re[i]);
    float ai = A_im[i];
    float dr = dt * ar, di = dt * ai;
    float ex = __expf(dr);
    float er = ex * cosf(di) - 1.0f;
    float ei = ex * sinf(di);
    float den = ar * ar + ai * ai;
    float qr = (er * ar + ei * ai) / den;
    float qi = (ei * ar - er * ai) / den;
    float cr = C_re[i], ci = C_im[i];
    Ct_re[i] = cr * qr - ci * qi;
    Ct_im[i] = cr * qi + ci * qr;
    dtA_re[i] = dr; dtA_im[i] = di;
}
__global__ void k_s4_K(const float* __restrict__ dtA_re, const float* __restrict__ dtA_im,
                       const float* __restrict__ Ct_re, const float* __restrict__ Ct_im,
                       float* __restrict__ Kmat) {
    int i = blockIdx.x * blockDim.x + threadIdx.x;   // d*L + l
    if (i >= HID_ * L_) return;
    int d = i / L_, l = i % L_;
    float s = 0.f;
    for (int n = 0; n < NST_; ++n) {
        int j = d * NST_ + n;
        float pr = dtA_re[j] * (float)l, pi = dtA_im[j] * (float)l;
        float m = __expf(pr);
        s += Ct_re[j] * (m * cosf(pi)) - Ct_im[j] * (m * sinf(pi));
    }
    Kmat[i] = 2.0f * s;
}
__global__ void k_s4_ksum(const float* __restrict__ Kmat, float* __restrict__ Ksum) {
    int d = blockIdx.x * blockDim.x + threadIdx.x;
    if (d >= HID_) return;
    float s = 0.f;
    for (int l = 0; l < L_; ++l) s += Kmat[d * L_ + l];
    Ksum[d] = s;
}
__global__ void k_krevT(const float* __restrict__ Kmat, bf16_t* __restrict__ KrevT) {
    int i = blockIdx.x * blockDim.x + threadIdx.x;   // t*HID + d
    if (i >= L_ * HID_) return;
    int t = i / HID_, d = i % HID_;
    KrevT[i] = (bf16_t)Kmat[d * L_ + (L_ - 1 - t)];
}

// ---------------- temporal epilogues (last position only; causal collapse) ----------------
__global__ void k_temporal_y(const float* __restrict__ x_temp, const float* __restrict__ conv_s,
                             const float* __restrict__ enc_w, const float* __restrict__ enc_b,
                             const float* __restrict__ Ksum, const float* __restrict__ s4D,
                             float* __restrict__ u_last, bf16_t* __restrict__ yact_bf) {
    int i = blockIdx.x * blockDim.x + threadIdx.x;   // b*HID + d
    if (i >= B_ * HID_) return;
    int b = i / HID_, d = i % HID_;
    float u = x_temp[b * L_ + (L_ - 1)] * enc_w[d] + enc_b[d];
    float conv = enc_w[d] * conv_s[i] + enc_b[d] * Ksum[d];
    float y = conv + u * s4D[d];
    u_last[i] = u;
    yact_bf[i] = (bf16_t)gelu_f(y);
}
__global__ void k_row_glu_ln(const float* __restrict__ glu_out, const float* __restrict__ glu_b,
                             const float* __restrict__ u_last, const float* __restrict__ g_,
                             const float* __restrict__ b_, bf16_t* __restrict__ h_bf) {
    __shared__ float red[128];
    int b = blockIdx.x, d = threadIdx.x;
    float a = glu_out[b * 256 + d] + glu_b[d];
    float g = glu_out[b * 256 + 128 + d] + glu_b[128 + d];
    float t = u_last[b * 128 + d] + a * sigmoid_f(g);
    h_bf[b * 128 + d] = (bf16_t)block_ln_128(t, g_[d], b_[d], red, d);
}
__global__ void k_temporal_out(const float* __restrict__ s4o, const float* __restrict__ dec_b,
                               const float* __restrict__ x_temp, const float* __restrict__ tcn_w,
                               const float* __restrict__ tcn_b, const float* __restrict__ g_,
                               const float* __restrict__ b_, float* __restrict__ tokens) {
    __shared__ float red[128];
    int b = blockIdx.x, d = threadIdx.x;
    // dilated causal TCN, dilation 2, kernel 3, pad (2,2): out[511] = w0*x[509] + w1*x[511]
    float tcn = tcn_w[d * 3 + 0] * x_temp[b * L_ + 509] + tcn_w[d * 3 + 1] * x_temp[b * L_ + 511] + tcn_b[d];
    float t = s4o[b * 128 + d] + dec_b[d] + tcn;
    tokens[b * 384 + d] = block_ln_128(t, g_[d], b_[d], red, d);   // token 0 = ht
}

// ---------------- graph branch ----------------
__global__ void k_in_proj(const float* __restrict__ coords, const float* __restrict__ W,
                          const float* __restrict__ bias, float* __restrict__ xg,
                          bf16_t* __restrict__ xg_bf) {
    int i = blockIdx.x * blockDim.x + threadIdx.x;   // n*HID + d
    if (i >= NNODE_ * HID_) return;
    int n = i / HID_, d = i % HID_;
    float v = coords[n * 2] * W[d] + coords[n * 2 + 1] * W[HID_ + d] + bias[d];
    xg[i] = v; xg_bf[i] = (bf16_t)v;
}
__global__ void k_gat_edge_e(const int* __restrict__ src, const int* __restrict__ dst,
                             const float* __restrict__ xl, const float* __restrict__ xr,
                             const float* __restrict__ att, float* __restrict__ e_buf,
                             unsigned* __restrict__ emax) {
    int t = blockIdx.x * blockDim.x + threadIdx.x;   // edge*2 + h
    if (t >= NEDGE_ * 2) return;
    int e = t >> 1, h = t & 1;
    const float* pl = xl + (size_t)src[e] * HID_ + h * 64;
    const float* pr = xr + (size_t)dst[e] * HID_ + h * 64;
    const float* pa = att + h * 64;
    float acc = 0.f;
    for (int c = 0; c < 64; ++c) {
        float m = pl[c] + pr[c];
        acc += (m > 0.f ? m : 0.2f * m) * pa[c];
    }
    e_buf[t] = acc;
    atomicMax(&emax[(size_t)dst[e] * 2 + h], fkey(acc));
}
__global__ void k_gat_edge_soft(const int* __restrict__ dst, const float* __restrict__ e_buf,
                                const unsigned* __restrict__ emax, float* __restrict__ ex_buf,
                                float* __restrict__ den) {
    int t = blockIdx.x * blockDim.x + threadIdx.x;
    if (t >= NEDGE_ * 2) return;
    int e = t >> 1, h = t & 1;
    int dd = dst[e];
    float ex = __expf(e_buf[t] - funkey(emax[(size_t)dd * 2 + h]));
    ex_buf[t] = ex;
    atomicAdd(&den[(size_t)dd * 2 + h], ex);
}
__global__ void k_gat_edge_agg(const int* __restrict__ src, const int* __restrict__ dst,
                               const float* __restrict__ xl, const float* __restrict__ ex_buf,
                               const float* __restrict__ den, float* __restrict__ acc) {
    int t = blockIdx.x * blockDim.x + threadIdx.x;   // edge*8 + part (2 heads x 4 chunks of 16)
    if (t >= NEDGE_ * 8) return;
    int e = t >> 3, part = t & 7;
    int h = part >> 2, c0 = (part & 3) * 16;
    int dd = dst[e];
    float alpha = ex_buf[e * 2 + h] / (den[(size_t)dd * 2 + h] + 1e-16f);
    const float* pl = xl + (size_t)src[e] * HID_ + h * 64 + c0;
    float* pa = acc + (size_t)dd * HID_ + h * 64 + c0;
    for (int c = 0; c < 16; ++c) atomicAdd(&pa[c], alpha * pl[c]);
}
__global__ void k_gat_node(const float* __restrict__ acc, const float* __restrict__ bias,
                           const float* __restrict__ g_, const float* __restrict__ b_,
                           float* __restrict__ xg, bf16_t* __restrict__ xg_bf) {
    __shared__ float red[128];
    int n = blockIdx.x, d = threadIdx.x;
    float v = acc[(size_t)n * HID_ + d] + bias[d];
    float el = v > 0.f ? v : (__expf(v) - 1.0f);               // ELU
    float t = el + xg[(size_t)n * HID_ + d];                   // residual
    float o = block_ln_128(t, g_[d], b_[d], red, d);
    xg[(size_t)n * HID_ + d] = o;
    xg_bf[(size_t)n * HID_ + d] = (bf16_t)o;
}
__global__ void k_gather_hg(const int* __restrict__ node_idx, const float* __restrict__ xg,
                            float* __restrict__ tokens) {
    int i = blockIdx.x * blockDim.x + threadIdx.x;
    if (i >= B_ * HID_) return;
    int b = i / HID_, d = i % HID_;
    tokens[b * 384 + 128 + d] = xg[(size_t)node_idx[b] * HID_ + d];   // token 1 = hg
}

// ---------------- extras / mixer / head epilogues ----------------
__global__ void k_bias_act_bf(const float* __restrict__ src, const float* __restrict__ bias,
                              bf16_t* __restrict__ dst, int total, int cols, int act) {
    int i = blockIdx.x * blockDim.x + threadIdx.x;
    if (i >= total) return;
    float v = src[i] + bias[i % cols];
    if (act == 1) v = v > 0.f ? v : 0.f;
    else if (act == 2) v = gelu_f(v);
    dst[i] = (bf16_t)v;
}
__global__ void k_he_out(const float* __restrict__ ex2, const float* __restrict__ b2,
                         float* __restrict__ tokens) {
    int i = blockIdx.x * blockDim.x + threadIdx.x;
    if (i >= B_ * HID_) return;
    int b = i / HID_, d = i % HID_;
    tokens[b * 384 + 256 + d] = ex2[i] + b2[d];                // token 2 = he
}
__global__ void k_attention(const float* __restrict__ qkv, const float* __restrict__ in_b,
                            float* __restrict__ o_buf) {
    int t = blockIdx.x * blockDim.x + threadIdx.x;             // b*2 + h
    if (t >= B_ * 2) return;
    int b = t >> 1, h = t & 1;
    float sc[3][3];
    for (int qt = 0; qt < 3; ++qt)
        for (int kt = 0; kt < 3; ++kt) {
            const float* q  = qkv + (size_t)(b * 3 + qt) * 384 + h * 64;
            const float* k  = qkv + (size_t)(b * 3 + kt) * 384 + 128 + h * 64;
            const float* qb = in_b + h * 64;
            const float* kb = in_b + 128 + h * 64;
            float s = 0.f;
            for (int c = 0; c < 64; ++c) s += (q[c] + qb[c]) * (k[c] + kb[c]);
            sc[qt][kt] = s * 0.125f;                            // 1/sqrt(64)
        }
    for (int qt = 0; qt < 3; ++qt) {
        float mx = fmaxf(sc[qt][0], fmaxf(sc[qt][1], sc[qt][2]));
        float e0 = __expf(sc[qt][0] - mx), e1 = __expf(sc[qt][1] - mx), e2 = __expf(sc[qt][2] - mx);
        float inv = 1.0f / (e0 + e1 + e2);
        float a0 = e0 * inv, a1 = e1 * inv, a2 = e2 * inv;
        const float* v0 = qkv + (size_t)(b * 3 + 0) * 384 + 256 + h * 64;
        const float* v1 = qkv + (size_t)(b * 3 + 1) * 384 + 256 + h * 64;
        const float* v2 = qkv + (size_t)(b * 3 + 2) * 384 + 256 + h * 64;
        const float* vb = in_b + 256 + h * 64;
        float* po = o_buf + (size_t)(b * 3 + qt) * 128 + h * 64;
        for (int c = 0; c < 64; ++c)
            po[c] = a0 * (v0[c] + vb[c]) + a1 * (v1[c] + vb[c]) + a2 * (v2[c] + vb[c]);
    }
}
__global__ void k_ln_add2(const float* __restrict__ base, const float* __restrict__ add,
                          const float* __restrict__ add_b, const float* __restrict__ g_,
                          const float* __restrict__ b_, float* __restrict__ out_f,
                          bf16_t* __restrict__ out_bf) {
    __shared__ float red[128];
    int r = blockIdx.x, d = threadIdx.x;
    float t = base[(size_t)r * 128 + d] + add[(size_t)r * 128 + d] + add_b[d];
    float o = block_ln_128(t, g_[d], b_[d], red, d);
    out_f[(size_t)r * 128 + d] = o;
    if (out_bf) out_bf[(size_t)r * 128 + d] = (bf16_t)o;
}
__global__ void k_pool(const float* __restrict__ x2, float* __restrict__ pooled,
                       bf16_t* __restrict__ pooled_bf) {
    int i = blockIdx.x * blockDim.x + threadIdx.x;
    if (i >= B_ * HID_) return;
    int b = i / HID_, d = i % HID_;
    float m = (x2[(b * 3 + 0) * 128 + d] + x2[(b * 3 + 1) * 128 + d] + x2[(b * 3 + 2) * 128 + d]) * (1.0f / 3.0f);
    pooled[i] = m; pooled_bf[i] = (bf16_t)m;
}
__global__ void k_head(const float* __restrict__ h1, const float* __restrict__ hd_b1,
                       const float* __restrict__ hd_w2, const float* __restrict__ hd_b2,
                       float* __restrict__ out) {
    int b = blockIdx.x * blockDim.x + threadIdx.x;
    if (b >= B_) return;
    float acc = hd_b2[0];
    for (int c = 0; c < 64; ++c) acc += gelu_f(h1[b * 64 + c] + hd_b1[c]) * hd_w2[c];
    out[b] = acc > 20.f ? acc : log1pf(__expf(acc));            // softplus
}

// ---------------- param index map (setup_inputs dict order, params nested in order) ----------------
enum {
    P_S4_ENC_W = 5, P_S4_ENC_B, P_S4_LOG_DT, P_S4_A_RE, P_S4_A_IM, P_S4_C_RE, P_S4_C_IM,
    P_S4_D, P_S4_GLU_W, P_S4_GLU_B, P_S4_LN_G, P_S4_LN_B, P_S4_DEC_W, P_S4_DEC_B,
    P_TCN_W, P_TCN_B, P_TEMP_LN_G, P_TEMP_LN_B, P_IN_PROJ_W, P_IN_PROJ_B,
    P_GAT0_WL, P_GAT0_WR, P_GAT0_ATT, P_GAT0_BIAS, P_GAT0_LN_G, P_GAT0_LN_B,
    P_GAT1_WL, P_GAT1_WR, P_GAT1_ATT, P_GAT1_BIAS, P_GAT1_LN_G, P_GAT1_LN_B,
    P_GAT2_WL, P_GAT2_WR, P_GAT2_ATT, P_GAT2_BIAS, P_GAT2_LN_G, P_GAT2_LN_B,
    P_EX_W1, P_EX_B1, P_EX_W2, P_EX_B2,
    P_MHA_IN_W, P_MHA_IN_B, P_MHA_OUT_W, P_MHA_OUT_B,
    P_LN1_G, P_LN1_B, P_LN2_G, P_LN2_B,
    P_FF_W1, P_FF_B1, P_FF_W2, P_FF_B2,
    P_HD_W1, P_HD_B1, P_HD_W2, P_HD_B2
};

extern "C" void kernel_launch(void* const* d_in, const int* in_sizes, int n_in,
                              void* d_out, int out_size, void* d_ws, size_t ws_size,
                              hipStream_t stream) {
    (void)in_sizes; (void)n_in; (void)out_size; (void)ws_size;
    const float* x_temp   = (const float*)d_in[0];
    const int*   node_idx = (const int*)d_in[1];
    const float* coords   = (const float*)d_in[2];
    const int*   ei       = (const int*)d_in[3];     // (2,E): src then dst
    const float* extras   = (const float*)d_in[4];
    const int* src = ei;
    const int* dst = ei + NEDGE_;
    auto F = [&](int i) { return (const float*)d_in[i]; };
    float* out = (float*)d_out;

    // deterministic workspace layout
    char* ws = (char*)d_ws;
    size_t off = 0;
    auto alloc = [&](size_t bytes) { size_t r = off; off += (bytes + 255) & ~(size_t)255; return r; };
    float*  dtA_re  = (float*)(ws + alloc(HID_ * NST_ * 4));
    float*  dtA_im  = (float*)(ws + alloc(HID_ * NST_ * 4));
    float*  Ct_re   = (float*)(ws + alloc(HID_ * NST_ * 4));
    float*  Ct_im   = (float*)(ws + alloc(HID_ * NST_ * 4));
    float*  Kmat    = (float*)(ws + alloc(HID_ * L_ * 4));
    float*  Ksum    = (float*)(ws + alloc(HID_ * 4));
    bf16_t* KrevT   = (bf16_t*)(ws + alloc(L_ * HID_ * 2));
    bf16_t* X_bf    = (bf16_t*)(ws + alloc(B_ * L_ * 2));
    float*  conv_s  = (float*)(ws + alloc(B_ * HID_ * 4));
    float*  u_last  = (float*)(ws + alloc(B_ * HID_ * 4));
    bf16_t* yact_bf = (bf16_t*)(ws + alloc(B_ * HID_ * 2));
    float*  glu_out = (float*)(ws + alloc(B_ * 256 * 4));
    bf16_t* h_bf    = (bf16_t*)(ws + alloc(B_ * HID_ * 2));
    float*  s4o     = (float*)(ws + alloc(B_ * HID_ * 4));
    float*  tokens  = (float*)(ws + alloc(B_ * 3 * HID_ * 4));
    bf16_t* tokens_bf = (bf16_t*)(ws + alloc(B_ * 3 * HID_ * 2));
    bf16_t* w_glu   = (bf16_t*)(ws + alloc(HID_ * 256 * 2));
    bf16_t* w_dec   = (bf16_t*)(ws + alloc(HID_ * HID_ * 2));
    bf16_t* w_gat   = (bf16_t*)(ws + alloc(6 * HID_ * HID_ * 2));   // wl0,wr0,wl1,wr1,wl2,wr2
    bf16_t* w_ex1   = (bf16_t*)(ws + alloc(32 * HID_ * 2));
    bf16_t* w_ex2   = (bf16_t*)(ws + alloc(HID_ * HID_ * 2));
    bf16_t* w_mi    = (bf16_t*)(ws + alloc(HID_ * 384 * 2));
    bf16_t* w_mo    = (bf16_t*)(ws + alloc(HID_ * HID_ * 2));
    bf16_t* w_f1    = (bf16_t*)(ws + alloc(HID_ * 256 * 2));
    bf16_t* w_f2    = (bf16_t*)(ws + alloc(256 * HID_ * 2));
    bf16_t* w_h1    = (bf16_t*)(ws + alloc(HID_ * 64 * 2));
    bf16_t* extras_bf = (bf16_t*)(ws + alloc(B_ * 32 * 2));
    float*  xg      = (float*)(ws + alloc((size_t)NNODE_ * HID_ * 4));
    bf16_t* xg_bf   = (bf16_t*)(ws + alloc((size_t)NNODE_ * HID_ * 2));
    float*  xl      = (float*)(ws + alloc((size_t)NNODE_ * HID_ * 4));
    float*  xr      = (float*)(ws + alloc((size_t)NNODE_ * HID_ * 4));
    float*  gacc    = (float*)(ws + alloc((size_t)NNODE_ * HID_ * 4));
    float*  e_buf   = (float*)(ws + alloc((size_t)NEDGE_ * 2 * 4));
    float*  ex_buf  = (float*)(ws + alloc((size_t)NEDGE_ * 2 * 4));
    unsigned* emax  = (unsigned*)(ws + alloc((size_t)NNODE_ * 2 * 4));
    float*  den     = (float*)(ws + alloc((size_t)NNODE_ * 2 * 4));
    float*  qkv     = (float*)(ws + alloc(1536 * 384 * 4));
    float*  o_buf   = (float*)(ws + alloc(1536 * 128 * 4));
    bf16_t* o_bf    = (bf16_t*)(ws + alloc(1536 * 128 * 2));
    float*  attno   = (float*)(ws + alloc(1536 * 128 * 4));
    float*  x1      = (float*)(ws + alloc(1536 * 128 * 4));
    bf16_t* x1_bf   = (bf16_t*)(ws + alloc(1536 * 128 * 2));
    float*  ff1     = (float*)(ws + alloc(1536 * 256 * 4));
    bf16_t* ff1_bf  = (bf16_t*)(ws + alloc(1536 * 256 * 2));
    float*  ff2     = (float*)(ws + alloc(1536 * 128 * 4));
    float*  x2      = (float*)(ws + alloc(1536 * 128 * 4));
    float*  pooled  = (float*)(ws + alloc(B_ * HID_ * 4));
    bf16_t* pooled_bf = (bf16_t*)(ws + alloc(B_ * HID_ * 2));
    float*  ex1buf  = (float*)(ws + alloc(B_ * HID_ * 4));
    bf16_t* ex1_bf  = (bf16_t*)(ws + alloc(B_ * HID_ * 2));
    float*  ex2buf  = (float*)(ws + alloc(B_ * HID_ * 4));
    float*  h1      = (float*)(ws + alloc(B_ * 64 * 4));

    auto cast = [&](const float* s, bf16_t* d, int n) {
        k_cast_bf16<<<(n + 255) / 256, 256, 0, stream>>>(s, d, n);
    };
    auto gemm = [&](const bf16_t* A, const bf16_t* Bm, float* C, int M, int N, int K) {
        k_gemm_bf16<<<dim3(M / 16, N / 16), dim3(32), 0, stream>>>(A, Bm, C, M, N, K);
    };

    // ---- weight/activation bf16 conversions ----
    cast(F(P_S4_GLU_W), w_glu, HID_ * 256);
    cast(F(P_S4_DEC_W), w_dec, HID_ * HID_);
    for (int l = 0; l < GL_; ++l) {
        cast(F(P_GAT0_WL + 6 * l), w_gat + (size_t)(2 * l) * HID_ * HID_, HID_ * HID_);
        cast(F(P_GAT0_WR + 6 * l), w_gat + (size_t)(2 * l + 1) * HID_ * HID_, HID_ * HID_);
    }
    cast(F(P_EX_W1), w_ex1, 32 * HID_);
    cast(F(P_EX_W2), w_ex2, HID_ * HID_);
    cast(F(P_MHA_IN_W), w_mi, HID_ * 384);
    cast(F(P_MHA_OUT_W), w_mo, HID_ * HID_);
    cast(F(P_FF_W1), w_f1, HID_ * 256);
    cast(F(P_FF_W2), w_f2, 256 * HID_);
    cast(F(P_HD_W1), w_h1, HID_ * 64);
    cast(extras, extras_bf, B_ * 32);
    cast(x_temp, X_bf, B_ * L_);

    // ---- temporal branch (only last position needed: causal collapse) ----
    k_s4_prep<<<(HID_ * NST_ + 255) / 256, 256, 0, stream>>>(F(P_S4_LOG_DT), F(P_S4_A_RE),
        F(P_S4_A_IM), F(P_S4_C_RE), F(P_S4_C_IM), dtA_re, dtA_im, Ct_re, Ct_im);
    k_s4_K<<<(HID_ * L_ + 255) / 256, 256, 0, stream>>>(dtA_re, dtA_im, Ct_re, Ct_im, Kmat);
    k_s4_ksum<<<1, 128, 0, stream>>>(Kmat, Ksum);
    k_krevT<<<(L_ * HID_ + 255) / 256, 256, 0, stream>>>(Kmat, KrevT);
    gemm(X_bf, KrevT, conv_s, B_, HID_, L_);                    // X @ Krev^T
    k_temporal_y<<<(B_ * HID_ + 255) / 256, 256, 0, stream>>>(x_temp, conv_s, F(P_S4_ENC_W),
        F(P_S4_ENC_B), Ksum, F(P_S4_D), u_last, yact_bf);
    gemm(yact_bf, w_glu, glu_out, B_, 256, HID_);
    k_row_glu_ln<<<B_, 128, 0, stream>>>(glu_out, F(P_S4_GLU_B), u_last, F(P_S4_LN_G), F(P_S4_LN_B), h_bf);
    gemm(h_bf, w_dec, s4o, B_, HID_, HID_);
    k_temporal_out<<<B_, 128, 0, stream>>>(s4o, F(P_S4_DEC_B), x_temp, F(P_TCN_W), F(P_TCN_B),
        F(P_TEMP_LN_G), F(P_TEMP_LN_B), tokens);

    // ---- graph branch ----
    k_in_proj<<<(NNODE_ * HID_ + 255) / 256, 256, 0, stream>>>(coords, F(P_IN_PROJ_W), F(P_IN_PROJ_B), xg, xg_bf);
    for (int l = 0; l < GL_; ++l) {
        gemm(xg_bf, w_gat + (size_t)(2 * l) * HID_ * HID_, xl, NNODE_, HID_, HID_);
        gemm(xg_bf, w_gat + (size_t)(2 * l + 1) * HID_ * HID_, xr, NNODE_, HID_, HID_);
        k_fill_u32<<<(NNODE_ * 2 + 255) / 256, 256, 0, stream>>>(emax, 0u, NNODE_ * 2);
        k_fill_u32<<<(NNODE_ * 2 + 255) / 256, 256, 0, stream>>>((unsigned*)den, 0u, NNODE_ * 2);
        k_gat_edge_e<<<(NEDGE_ * 2 + 255) / 256, 256, 0, stream>>>(src, dst, xl, xr,
            F(P_GAT0_ATT + 6 * l), e_buf, emax);
        k_gat_edge_soft<<<(NEDGE_ * 2 + 255) / 256, 256, 0, stream>>>(dst, e_buf, emax, ex_buf, den);
        k_fill_u32<<<(NNODE_ * HID_ + 255) / 256, 256, 0, stream>>>((unsigned*)gacc, 0u, NNODE_ * HID_);
        k_gat_edge_agg<<<(NEDGE_ * 8 + 255) / 256, 256, 0, stream>>>(src, dst, xl, ex_buf, den, gacc);
        k_gat_node<<<NNODE_, 128, 0, stream>>>(gacc, F(P_GAT0_BIAS + 6 * l),
            F(P_GAT0_LN_G + 6 * l), F(P_GAT0_LN_B + 6 * l), xg, xg_bf);
    }
    k_gather_hg<<<(B_ * HID_ + 255) / 256, 256, 0, stream>>>(node_idx, xg, tokens);

    // ---- extras branch ----
    gemm(extras_bf, w_ex1, ex1buf, B_, HID_, 32);
    k_bias_act_bf<<<(B_ * HID_ + 255) / 256, 256, 0, stream>>>(ex1buf, F(P_EX_B1), ex1_bf, B_ * HID_, HID_, 1);
    gemm(ex1_bf, w_ex2, ex2buf, B_, HID_, HID_);
    k_he_out<<<(B_ * HID_ + 255) / 256, 256, 0, stream>>>(ex2buf, F(P_EX_B2), tokens);

    // ---- mixer ----
    cast(tokens, tokens_bf, B_ * 3 * HID_);
    gemm(tokens_bf, w_mi, qkv, B_ * 3, 384, HID_);
    k_attention<<<(B_ * 2 + 255) / 256, 256, 0, stream>>>(qkv, F(P_MHA_IN_B), o_buf);
    cast(o_buf, o_bf, B_ * 3 * HID_);
    gemm(o_bf, w_mo, attno, B_ * 3, HID_, HID_);
    k_ln_add2<<<B_ * 3, 128, 0, stream>>>(tokens, attno, F(P_MHA_OUT_B), F(P_LN1_G), F(P_LN1_B), x1, x1_bf);
    gemm(x1_bf, w_f1, ff1, B_ * 3, 256, HID_);
    k_bias_act_bf<<<(B_ * 3 * 256 + 255) / 256, 256, 0, stream>>>(ff1, F(P_FF_B1), ff1_bf, B_ * 3 * 256, 256, 2);
    gemm(ff1_bf, w_f2, ff2, B_ * 3, HID_, 256);
    k_ln_add2<<<B_ * 3, 128, 0, stream>>>(x1, ff2, F(P_FF_B2), F(P_LN2_G), F(P_LN2_B), x2, (bf16_t*)nullptr);
    k_pool<<<(B_ * HID_ + 255) / 256, 256, 0, stream>>>(x2, pooled, pooled_bf);

    // ---- head ----
    gemm(pooled_bf, w_h1, h1, B_, 64, HID_);
    k_head<<<(B_ + 255) / 256, 256, 0, stream>>>(h1, F(P_HD_B1), F(P_HD_W2), F(P_HD_B2), out);
}
